// SingleHeadAttention_57449482551634
// MI455X (gfx1250) — compile-verified, matrix operand order checked by reading, not run on hardware
//
#include <hip/hip_runtime.h>
#include <hip/hip_bf16.h>

// ---------------------------------------------------------------------------
// Types for CDNA5 WMMA (wave32): v_wmma_f32_16x16x32_bf16
// ---------------------------------------------------------------------------
typedef __attribute__((ext_vector_type(16))) __bf16          v16bf;
typedef __attribute__((ext_vector_type(8)))  float           v8f;
typedef __attribute__((ext_vector_type(4)))  float           f32x4;
typedef __attribute__((ext_vector_type(4)))  unsigned short  u16x4;
typedef __attribute__((ext_vector_type(8)))  unsigned short  u16x8;
typedef __attribute__((ext_vector_type(16))) unsigned short  u16x16;

// Problem constants (match the reference)
#define B_BATCH 4
#define SEQ     2048
#define DMODEL  1024
#define DKQ     1024
#define MTOT    (B_BATCH * SEQ)     // 8192 flattened rows for projections
#define SCALE_QK 0.03125f           // 1/sqrt(1024)

#define APAD 40                     // padded LDS row stride (elems); 80B, 16B-aligned

// ---------------------------------------------------------------------------
// Helpers
// ---------------------------------------------------------------------------
__device__ __forceinline__ unsigned short cvt_bf16(float f) {
  unsigned int u = __builtin_bit_cast(unsigned int, f);
  u += 0x7FFFu + ((u >> 16) & 1u);          // round-to-nearest-even
  return (unsigned short)(u >> 16);
}

__device__ __forceinline__ v8f wmma_bf16(v16bf a, v16bf b, v8f c) {
  return __builtin_amdgcn_wmma_f32_16x16x32_bf16(false, a, false, b,
                                                 (short)0, c, false, false);
}

// Build one 16x32 bf16 fragment from an LDS row (K-contiguous, padded APAD).
// ISA layout: lanes 0-15 row=lane, elems 0..7 = K+0..7, elems 8..15 = K+16..23;
// lanes 16-31 shifted by +8 in K.
__device__ __forceinline__ v16bf frag_from_lds(const unsigned short* r) {
  u16x8 lo = *(const u16x8*)(r);
  u16x8 hi = *(const u16x8*)(r + 16);
  u16x16 t;
#pragma unroll
  for (int i = 0; i < 8; ++i) { t[i] = lo[i]; t[i + 8] = hi[i]; }
  return __builtin_bit_cast(v16bf, t);
}

// ---------------------------------------------------------------------------
// Shared GEMM mainloop: block tile 128(M) x 256(N), 8 waves of 64x64.
// A: [128 rows x ldA], B: [256 rows x ldB], both bf16 K-contiguous rows.
// Tiles staged into LDS with GLOBAL_LOAD_ASYNC_TO_LDS_B128 (ASYNCcnt),
// double-buffered so the DMA of tile k+1 overlaps the WMMAs of tile k.
// ---------------------------------------------------------------------------
__device__ __forceinline__ void gemm_mainloop(const unsigned short* __restrict__ Abase,
                                              size_t ldA,
                                              const unsigned short* __restrict__ Bbase,
                                              size_t ldB,
                                              int K, v8f acc[4][4]) {
  __shared__ unsigned short sA[2][128 * APAD];   // 2 x 10.0 KB
  __shared__ unsigned short sB[2][256 * APAD];   // 2 x 20.0 KB
  const int t = threadIdx.x;
  const int lane = t & 31;
  const int wave = t >> 5;
  const int wm = (wave & 1) * 64;    // wave M offset within block tile
  const int wn = (wave >> 1) * 64;   // wave N offset within block tile
  const int row_a = lane & 15;
  const int h8 = ((lane >> 4) & 1) * 8;

  // Cooperative async stage of one 32-wide K slice of A (128 rows) and B (256
  // rows) into LDS buffer p. 16B chunks; 512 A-chunks + 1024 B-chunks over
  // 256 threads -> 2 + 4 async b128 ops per thread.
  auto stage = [&](int k0, int p) {
#pragma unroll
    for (int u = 0; u < 2; ++u) {
      const int ch = t + u * 256;
      const int row = ch >> 2, sub = ch & 3;
      const unsigned short* g = Abase + (size_t)row * ldA + k0 + sub * 8;
      const unsigned lds = (unsigned)(size_t)&sA[p][row * APAD + sub * 8];
      asm volatile("global_load_async_to_lds_b128 %0, %1, off"
                   :: "v"(lds), "v"((unsigned long long)(size_t)g) : "memory");
    }
#pragma unroll
    for (int u = 0; u < 4; ++u) {
      const int ch = t + u * 256;
      const int row = ch >> 2, sub = ch & 3;
      const unsigned short* g = Bbase + (size_t)row * ldB + k0 + sub * 8;
      const unsigned lds = (unsigned)(size_t)&sB[p][row * APAD + sub * 8];
      asm volatile("global_load_async_to_lds_b128 %0, %1, off"
                   :: "v"(lds), "v"((unsigned long long)(size_t)g) : "memory");
    }
  };

  stage(0, 0);
  asm volatile("s_wait_asynccnt 0" ::: "memory");
  __syncthreads();

  int p = 0;
  for (int k0 = 0; k0 < K; k0 += 32) {
    const bool more = (k0 + 32) < K;
    if (more) stage(k0 + 32, p ^ 1);    // async DMA overlaps compute below

    v16bf a[4], b[4];
#pragma unroll
    for (int i = 0; i < 4; ++i)
      a[i] = frag_from_lds(&sA[p][(wm + 16 * i + row_a) * APAD + h8]);
#pragma unroll
    for (int j = 0; j < 4; ++j)
      b[j] = frag_from_lds(&sB[p][(wn + 16 * j + row_a) * APAD + h8]);
#pragma unroll
    for (int i = 0; i < 4; ++i)
#pragma unroll
      for (int j = 0; j < 4; ++j)
        acc[i][j] = wmma_bf16(a[i], b[j], acc[i][j]);

    if (more) {
      asm volatile("s_wait_asynccnt 0" ::: "memory");  // next tile landed
      __syncthreads();  // all waves done reading buf p / writes visible
    }
    p ^= 1;
  }
}

// ---------------------------------------------------------------------------
// Kernel 0: bulk f32 -> bf16 conversion (bandwidth bound, done once per tensor)
// n must be a multiple of 1024; grid = n/1024, block = 256, 4 elems/thread.
// ---------------------------------------------------------------------------
__global__ void __launch_bounds__(256)
cvt_f32_bf16_kernel(const float* __restrict__ src, unsigned short* __restrict__ dst) {
  const size_t i = ((size_t)blockIdx.x * 256 + threadIdx.x) * 4;
  f32x4 v = *(const f32x4*)(src + i);
  u16x4 o;
#pragma unroll
  for (int j = 0; j < 4; ++j) o[j] = cvt_bf16(v[j]);
  *(u16x4*)(dst + i) = o;
}

// ---------------------------------------------------------------------------
// Kernel 1: projection GEMM  Y[m,n] = sum_k X[m,k] * W[n,k]   (y = x @ W^T)
//   X: bf16 [MTOT x K], W: bf16 [N x K].
//   Y: bf16. TRANS=0 -> Y[m*ldy + n]; TRANS=1 -> Y[n*ldy + m] (for V^T).
// Grid = (M/128, N/256), block 256.
// ---------------------------------------------------------------------------
template <int TRANS>
__global__ void __launch_bounds__(256)
proj_gemm_kernel(const unsigned short* __restrict__ X,
                 const unsigned short* __restrict__ W,
                 unsigned short* __restrict__ Y, int K, int ldy) {
  const int lane = threadIdx.x & 31;
  const int wave = threadIdx.x >> 5;
  const int m0 = blockIdx.x * 128 + (wave & 1) * 64;
  const int n0 = blockIdx.y * 256 + (wave >> 1) * 64;

  v8f acc[4][4] = {};
  gemm_mainloop(X + (size_t)blockIdx.x * 128 * K, K,
                W + (size_t)blockIdx.y * 256 * K, K, K, acc);

  // C/D layout: VGPR r -> M = r + 8*(lane>=16), N = lane&15
  const int mh = (lane >> 4) * 8;
  const int nn = lane & 15;
#pragma unroll
  for (int i = 0; i < 4; ++i)
#pragma unroll
    for (int j = 0; j < 4; ++j) {
      const int mbase = m0 + 16 * i + mh;
      const int ncol  = n0 + 16 * j + nn;
      if (TRANS) {
        u16x8 tv;
#pragma unroll
        for (int r = 0; r < 8; ++r) tv[r] = cvt_bf16(acc[i][j][r]);
        *(u16x8*)(Y + (size_t)ncol * ldy + mbase) = tv;   // 16B contiguous
      } else {
#pragma unroll
        for (int r = 0; r < 8; ++r)
          Y[(size_t)(mbase + r) * ldy + ncol] = cvt_bf16(acc[i][j][r]);
      }
    }
}

// ---------------------------------------------------------------------------
// Kernel 2: scores[b,i,j] = (q[b,i,:] . k[b,j,:]) * 1/sqrt(Dkq)   (f32 out)
//   q,k: bf16 [MTOT x DKQ].  Grid (16,8,4), block 256.
// ---------------------------------------------------------------------------
__global__ void __launch_bounds__(256)
scores_gemm_kernel(const unsigned short* __restrict__ q,
                   const unsigned short* __restrict__ k,
                   float* __restrict__ scores) {
  const int lane = threadIdx.x & 31;
  const int wave = threadIdx.x >> 5;
  const int b = blockIdx.z;
  const int m0 = blockIdx.x * 128 + (wave & 1) * 64;
  const int n0 = blockIdx.y * 256 + (wave >> 1) * 64;

  v8f acc[4][4] = {};
  gemm_mainloop(q + ((size_t)b * SEQ + blockIdx.x * 128) * DKQ, DKQ,
                k + ((size_t)b * SEQ + blockIdx.y * 256) * DKQ, DKQ, DKQ, acc);

  float* sb = scores + (size_t)b * SEQ * SEQ;
  const int mh = (lane >> 4) * 8;
  const int nn = lane & 15;
#pragma unroll
  for (int i = 0; i < 4; ++i)
#pragma unroll
    for (int j = 0; j < 4; ++j) {
      const int mbase = m0 + 16 * i + mh;
      const int ncol  = n0 + 16 * j + nn;
#pragma unroll
      for (int r = 0; r < 8; ++r)
        sb[(size_t)(mbase + r) * SEQ + ncol] = acc[i][j][r] * SCALE_QK;
    }
}

// ---------------------------------------------------------------------------
// Kernel 3: row softmax over SEQ=2048 f32 scores; emits bf16 attn.
// One block (256 threads) per row; skips the f32 write-back entirely.
// ---------------------------------------------------------------------------
__global__ void __launch_bounds__(256)
softmax_kernel(const float* __restrict__ scores, unsigned short* __restrict__ attn) {
  __shared__ float red[256];
  const float* p = scores + (size_t)blockIdx.x * SEQ;
  unsigned short* o = attn + (size_t)blockIdx.x * SEQ;
  const int t = threadIdx.x;

  float loc[8];
  float mx = -3.0e38f;
#pragma unroll
  for (int i = 0; i < 8; ++i) {
    loc[i] = p[t + 256 * i];
    mx = fmaxf(mx, loc[i]);
  }
  red[t] = mx; __syncthreads();
  for (int s = 128; s > 0; s >>= 1) {
    if (t < s) red[t] = fmaxf(red[t], red[t + s]);
    __syncthreads();
  }
  mx = red[0]; __syncthreads();

  float sum = 0.0f;
#pragma unroll
  for (int i = 0; i < 8; ++i) {
    loc[i] = __expf(loc[i] - mx);
    sum += loc[i];
  }
  red[t] = sum; __syncthreads();
  for (int s = 128; s > 0; s >>= 1) {
    if (t < s) red[t] += red[t + s];
    __syncthreads();
  }
  const float inv = 1.0f / red[0];
#pragma unroll
  for (int i = 0; i < 8; ++i) o[t + 256 * i] = cvt_bf16(loc[i] * inv);
}

// ---------------------------------------------------------------------------
// Kernel 4: out[b,i,e] = sum_j attn[b,i,j] * v[b,j,e]
//   attn: bf16 [SEQ x SEQ] per batch (K-contiguous)
//   vT:   bf16 [DMODEL x MTOT] (v stored transposed -> K-contiguous rows)
//   out:  f32 [B x SEQ x DMODEL].  Grid (16,4,4), block 256.
// ---------------------------------------------------------------------------
__global__ void __launch_bounds__(256)
out_gemm_kernel(const unsigned short* __restrict__ attn,
                const unsigned short* __restrict__ vT,
                float* __restrict__ out) {
  const int lane = threadIdx.x & 31;
  const int wave = threadIdx.x >> 5;
  const int b = blockIdx.z;
  const int m0 = blockIdx.x * 128 + (wave & 1) * 64;
  const int n0 = blockIdx.y * 256 + (wave >> 1) * 64;

  v8f acc[4][4] = {};
  gemm_mainloop(attn + (size_t)b * SEQ * SEQ + (size_t)blockIdx.x * 128 * SEQ, SEQ,
                vT + (size_t)blockIdx.y * 256 * MTOT + (size_t)b * SEQ, MTOT,
                SEQ, acc);

  float* ob = out + (size_t)b * SEQ * DMODEL;
  const int mh = (lane >> 4) * 8;
  const int nn = lane & 15;
#pragma unroll
  for (int i = 0; i < 4; ++i)
#pragma unroll
    for (int j = 0; j < 4; ++j) {
      const int mbase = m0 + 16 * i + mh;
      const int ncol  = n0 + 16 * j + nn;
#pragma unroll
      for (int r = 0; r < 8; ++r)
        ob[(size_t)(mbase + r) * DMODEL + ncol] = acc[i][j][r];
    }
}

// ---------------------------------------------------------------------------
// Launch
// ---------------------------------------------------------------------------
extern "C" void kernel_launch(void* const* d_in, const int* in_sizes, int n_in,
                              void* d_out, int out_size, void* d_ws, size_t ws_size,
                              hipStream_t stream) {
  (void)in_sizes; (void)n_in; (void)out_size; (void)ws_size;
  const float* x  = (const float*)d_in[0];
  const float* Wq = (const float*)d_in[1];
  const float* Wk = (const float*)d_in[2];
  const float* Wv = (const float*)d_in[3];
  float* out = (float*)d_out;

  char* ws = (char*)d_ws;
  const size_t MB = (size_t)1 << 20;
  unsigned short* xb   = (unsigned short*)(ws);              // 16 MB
  unsigned short* Wqb  = (unsigned short*)(ws + 16 * MB);    //  2 MB
  unsigned short* Wkb  = (unsigned short*)(ws + 18 * MB);    //  2 MB
  unsigned short* Wvb  = (unsigned short*)(ws + 20 * MB);    //  2 MB
  unsigned short* qb   = (unsigned short*)(ws + 22 * MB);    // 16 MB
  unsigned short* kb   = (unsigned short*)(ws + 38 * MB);    // 16 MB
  unsigned short* vTb  = (unsigned short*)(ws + 54 * MB);    // 16 MB
  float*          scor = (float*)(ws + 70 * MB);             // 64 MB
  unsigned short* attn = (unsigned short*)(ws + 134 * MB);   // 32 MB

  dim3 blk(256);

  // One-time f32 -> bf16 conversions (removes all cvt work from GEMM loops)
  cvt_f32_bf16_kernel<<<dim3((MTOT * DMODEL) / 1024), blk, 0, stream>>>(x, xb);
  cvt_f32_bf16_kernel<<<dim3((DKQ * DMODEL) / 1024), blk, 0, stream>>>(Wq, Wqb);
  cvt_f32_bf16_kernel<<<dim3((DKQ * DMODEL) / 1024), blk, 0, stream>>>(Wk, Wkb);
  cvt_f32_bf16_kernel<<<dim3((DMODEL * DMODEL) / 1024), blk, 0, stream>>>(Wv, Wvb);

  // Projections: q = x Wq^T, k = x Wk^T (row-major bf16); v stored transposed
  proj_gemm_kernel<0><<<dim3(MTOT / 128, DKQ / 256, 1), blk, 0, stream>>>(
      xb, Wqb, qb, DMODEL, DKQ);
  proj_gemm_kernel<0><<<dim3(MTOT / 128, DKQ / 256, 1), blk, 0, stream>>>(
      xb, Wkb, kb, DMODEL, DKQ);
  proj_gemm_kernel<1><<<dim3(MTOT / 128, DMODEL / 256, 1), blk, 0, stream>>>(
      xb, Wvb, vTb, DMODEL, MTOT);

  // scores = (q k^T) / sqrt(Dkq)
  scores_gemm_kernel<<<dim3(SEQ / 128, SEQ / 256, B_BATCH), blk, 0, stream>>>(
      qb, kb, scor);

  // softmax rows -> bf16 attention weights
  softmax_kernel<<<dim3(B_BATCH * SEQ), blk, 0, stream>>>(scor, attn);

  // out = attn @ v
  out_gemm_kernel<<<dim3(SEQ / 128, DMODEL / 256, B_BATCH), blk, 0, stream>>>(
      attn, vTb, out);
}
